// SelfAttention_84018150244637
// MI455X (gfx1250) — matrix operand order repaired and checked by reading, not compiled, not run on hardware
//
#include <hip/hip_runtime.h>
#include <hip/hip_bf16.h>

// ---------------------------------------------------------------------------
// Transformer-XL relative self-attention for MI455X (gfx1250, wave32, WMMA).
// All heavy math on v_wmma_f32_16x16x32_bf16 (bf16 in, f32 accumulate).
// Flash-attention inner loop keeps the 285MB logit tensor entirely in LDS.
// ---------------------------------------------------------------------------

typedef __attribute__((ext_vector_type(16))) __bf16 v16bf;
typedef __attribute__((ext_vector_type(8)))  __bf16 v8bf;
typedef __attribute__((ext_vector_type(4)))  __bf16 v4bf;
typedef __attribute__((ext_vector_type(8)))  float  v8f;

#define BB     2
#define TT     1024
#define EE     1024
#define HH     16
#define DHD    64
#define KVLEN  2180
#define KVPAD  2240   // 35 * 64
#define PEPAD  2304   // covers jj up to 2303
#define TMEM   1156   // kv_len - t

__device__ __forceinline__ __bf16 f2bf(float f) {
  unsigned u = __float_as_uint(f);
  u += 0x7fffu + ((u >> 16) & 1u);            // round-to-nearest-even
  unsigned short s = (unsigned short)(u >> 16);
  __bf16 r;
  __builtin_memcpy(&r, &s, 2);
  return r;
}

// A fragment (16x32 bf16, M across lanes): lane m=L%16 reads K = 8*half..+7 and 16+8*half..+7
__device__ __forceinline__ v16bf frag_a(const __bf16* tile, int ld) {
  int lane = threadIdx.x & 31;
  int row  = lane & 15;
  int hf   = lane >> 4;
  const __bf16* p = tile + row * ld + 8 * hf;
  v8bf lo = *(const v8bf*)p;
  v8bf hi = *(const v8bf*)(p + 16);
  return __builtin_shufflevector(lo, hi, 0,1,2,3,4,5,6,7,8,9,10,11,12,13,14,15);
}

// B fragment (32x16 bf16, N across lanes): lane n=L%16 reads K = 8*half..+7 and 16+8*half..+7
__device__ __forceinline__ v16bf frag_b(const __bf16* tile, int ld) {
  int lane = threadIdx.x & 31;
  int col  = lane & 15;
  int hf   = lane >> 4;
  const __bf16* p = tile + col * ld + 8 * hf;
  v8bf lo = *(const v8bf*)p;
  v8bf hi = *(const v8bf*)(p + 16);
  return __builtin_shufflevector(lo, hi, 0,1,2,3,4,5,6,7,8,9,10,11,12,13,14,15);
}

__device__ __forceinline__ v8f wmma_bf16(v16bf a, v16bf b, v8f c) {
  return __builtin_amdgcn_wmma_f32_16x16x32_bf16(false, a, false, b, (short)0, c, false, false);
}

// ---------------------------------------------------------------------------
// pos_emb f32 [16,2180,64] -> bf16 [16,2304,64] zero padded
// ---------------------------------------------------------------------------
__global__ void __launch_bounds__(256) pe_pad_kernel(const float* __restrict__ pe,
                                                     __bf16* __restrict__ pebf) {
  int idx = blockIdx.x * 256 + threadIdx.x;
  if (idx >= HH * PEPAD * DHD) return;
  int d  = idx & 63;
  int jj = (idx >> 6) % PEPAD;
  int h  = idx / (PEPAD * DHD);
  pebf[idx] = (jj < KVLEN) ? f2bf(pe[((size_t)h * KVLEN + jj) * DHD + d]) : f2bf(0.f);
}

// zero K/V pad rows [2180, 2240)
__global__ void __launch_bounds__(256) kv_pad_kernel(__bf16* __restrict__ kbf,
                                                     __bf16* __restrict__ vbf) {
  int idx = blockIdx.x * 256 + threadIdx.x;
  const int PADR = KVPAD - KVLEN;  // 60
  if (idx >= BB * HH * PADR * DHD) return;
  int d  = idx & 63;
  int r  = (idx >> 6) % PADR;
  int bh = idx / (PADR * DHD);
  size_t off = ((size_t)bh * KVPAD + KVLEN + r) * DHD + d;
  kbf[off] = f2bf(0.f);
  vbf[off] = f2bf(0.f);
}

// ---------------------------------------------------------------------------
// Q projection: Q = X[2048,1024] @ Wq^T[1024,1024], scattered to [b,h,t,dh] bf16
// ---------------------------------------------------------------------------
__global__ void __launch_bounds__(256) proj_q_kernel(const float* __restrict__ x,
                                                     const float* __restrict__ Wq,
                                                     __bf16* __restrict__ qbf) {
  __shared__ __bf16 sA[64 * 40];
  __shared__ __bf16 sB[64 * 40];
  int tid = threadIdx.x;
  int m0 = blockIdx.x * 64, n0 = blockIdx.y * 64;
  int wv = tid >> 5, lane = tid & 31, hf = lane >> 4, col = lane & 15;
  int i0 = wv * 2, i1 = i0 + 1;
  v8f acc0 = {0.f,0.f,0.f,0.f,0.f,0.f,0.f,0.f};
  v8f acc1 = acc0;
  for (int k0 = 0; k0 < EE; k0 += 32) {
    for (int i = tid; i < 512; i += 256) {
      int r = i >> 3, c = (i & 7) << 2;
      float4 fa = *(const float4*)(x  + (size_t)(m0 + r) * EE + k0 + c);
      float4 fb = *(const float4*)(Wq + (size_t)(n0 + r) * EE + k0 + c);
      __bf16* da = sA + r * 40 + c;
      da[0] = f2bf(fa.x); da[1] = f2bf(fa.y); da[2] = f2bf(fa.z); da[3] = f2bf(fa.w);
      __bf16* db = sB + r * 40 + c;
      db[0] = f2bf(fb.x); db[1] = f2bf(fb.y); db[2] = f2bf(fb.z); db[3] = f2bf(fb.w);
    }
    __syncthreads();
    acc0 = wmma_bf16(frag_a(sA + (i0 >> 2) * 16 * 40, 40), frag_b(sB + (i0 & 3) * 16 * 40, 40), acc0);
    acc1 = wmma_bf16(frag_a(sA + (i1 >> 2) * 16 * 40, 40), frag_b(sB + (i1 & 3) * 16 * 40, 40), acc1);
    __syncthreads();
  }
#pragma unroll
  for (int r = 0; r < 8; r++) {
    {
      int mg = m0 + (i0 >> 2) * 16 + r + 8 * hf;
      int ng = n0 + (i0 & 3) * 16 + col;
      qbf[(((size_t)(mg >> 10) * HH + (ng >> 6)) * TT + (mg & 1023)) * DHD + (ng & 63)] = f2bf(acc0[r]);
    }
    {
      int mg = m0 + (i1 >> 2) * 16 + r + 8 * hf;
      int ng = n0 + (i1 & 3) * 16 + col;
      qbf[(((size_t)(mg >> 10) * HH + (ng >> 6)) * TT + (mg & 1023)) * DHD + (ng & 63)] = f2bf(acc1[r]);
    }
  }
}

// ---------------------------------------------------------------------------
// KV projection: gathers virtual concat [mem_kv | lmem | mem | x] rows,
// C[4360, 2048] = KVin @ Wkv^T, split/scattered to K/V [b,h,2240,64] bf16
// ---------------------------------------------------------------------------
__global__ void __launch_bounds__(256) proj_kv_kernel(const float* __restrict__ x,
                                                      const float* __restrict__ memp,
                                                      const float* __restrict__ lmem,
                                                      const float* __restrict__ mem_kv,
                                                      const float* __restrict__ Wkv,
                                                      __bf16* __restrict__ kbf,
                                                      __bf16* __restrict__ vbf) {
  __shared__ __bf16 sA[64 * 40];
  __shared__ __bf16 sB[64 * 40];
  int tid = threadIdx.x;
  int m0 = blockIdx.x * 64, n0 = blockIdx.y * 64;
  int wv = tid >> 5, lane = tid & 31, hf = lane >> 4, col = lane & 15;
  int i0 = wv * 2, i1 = i0 + 1;
  v8f acc0 = {0.f,0.f,0.f,0.f,0.f,0.f,0.f,0.f};
  v8f acc1 = acc0;
  for (int k0 = 0; k0 < EE; k0 += 32) {
    for (int i = tid; i < 512; i += 256) {
      int r = i >> 3, c = (i & 7) << 2;
      int rg = m0 + r;
      float4 fa = make_float4(0.f, 0.f, 0.f, 0.f);
      if (rg < BB * KVLEN) {
        int bb = (rg >= KVLEN) ? 1 : 0;
        int rr = rg - bb * KVLEN;
        const float* src;
        if (rr < 4)          src = mem_kv + (size_t)rr * EE;
        else if (rr < 132)   src = lmem + ((size_t)bb * 128  + (rr - 4))    * EE;
        else if (rr < TMEM)  src = memp + ((size_t)bb * 1024 + (rr - 132))  * EE;
        else                 src = x    + ((size_t)bb * TT   + (rr - TMEM)) * EE;
        fa = *(const float4*)(src + k0 + c);
      }
      float4 fb = *(const float4*)(Wkv + (size_t)(n0 + r) * EE + k0 + c);
      __bf16* da = sA + r * 40 + c;
      da[0] = f2bf(fa.x); da[1] = f2bf(fa.y); da[2] = f2bf(fa.z); da[3] = f2bf(fa.w);
      __bf16* db = sB + r * 40 + c;
      db[0] = f2bf(fb.x); db[1] = f2bf(fb.y); db[2] = f2bf(fb.z); db[3] = f2bf(fb.w);
    }
    __syncthreads();
    acc0 = wmma_bf16(frag_a(sA + (i0 >> 2) * 16 * 40, 40), frag_b(sB + (i0 & 3) * 16 * 40, 40), acc0);
    acc1 = wmma_bf16(frag_a(sA + (i1 >> 2) * 16 * 40, 40), frag_b(sB + (i1 & 3) * 16 * 40, 40), acc1);
    __syncthreads();
  }
#pragma unroll
  for (int r = 0; r < 8; r++) {
#pragma unroll
    for (int j = 0; j < 2; j++) {
      int ti = j ? i1 : i0;
      float av = j ? acc1[r] : acc0[r];
      int mg = m0 + (ti >> 2) * 16 + r + 8 * hf;
      int ng = n0 + (ti & 3) * 16 + col;
      if (mg < BB * KVLEN) {
        int bb = (mg >= KVLEN) ? 1 : 0;
        int rr = mg - bb * KVLEN;
        __bf16 val = f2bf(av);
        if (ng < EE)
          kbf[(((size_t)bb * HH + (ng >> 6)) * KVPAD + rr) * DHD + (ng & 63)] = val;
        else {
          int n2 = ng - EE;
          vbf[(((size_t)bb * HH + (n2 >> 6)) * KVPAD + rr) * DHD + (n2 & 63)] = val;
        }
      }
    }
  }
}

// ---------------------------------------------------------------------------
// Flash attention: one workgroup = 64 query rows of one (b,h).
// S = Q K^T, R = Q PE^T (128-wide strip), logit = scale*(S + R[shifted]),
// online softmax, O += P V with WMMA. LDS-resident throughout.
// ---------------------------------------------------------------------------
__global__ void __launch_bounds__(256) flash_attn_kernel(const __bf16* __restrict__ qbf,
                                                         const __bf16* __restrict__ kbf,
                                                         const __bf16* __restrict__ vbf,
                                                         const __bf16* __restrict__ pebf,
                                                         const unsigned char* __restrict__ im,
                                                         __bf16* __restrict__ aout) {
  extern __shared__ char smem[];
  __bf16* sQ  = (__bf16*)smem;       // 64 x 80
  __bf16* sK  = sQ  + 64 * 80;       // 64 x 80
  __bf16* sVt = sK  + 64 * 80;       // 64(d) x 80(key)
  __bf16* sPE = sVt + 64 * 80;       // 128 x 80
  __bf16* sP  = sPE + 128 * 80;      // 64 x 80
  float*  sS  = (float*)(sP + 64 * 80);   // 64 x 64
  float*  sR  = sS + 64 * 64;             // 64 x 128
  float*  sAl = sR + 64 * 128;            // 64
  float*  sL  = sAl + 64;                 // 64

  int tid = threadIdx.x;
  int m0 = blockIdx.x * 64;
  int bh = blockIdx.y;
  int b = bh >> 4, h = bh & 15;
  int wv = tid >> 5, lane = tid & 31, hf = lane >> 4, lcol = lane & 15;

  const __bf16* qbase  = qbf  + ((size_t)bh * TT + m0) * DHD;
  const __bf16* kbase  = kbf  + (size_t)bh * KVPAD * DHD;
  const __bf16* vbase  = vbf  + (size_t)bh * KVPAD * DHD;
  const __bf16* pebase = pebf + (size_t)h * PEPAD * DHD;

  for (int i = tid; i < 64 * 16; i += 256) {
    int r = i >> 4, c = (i & 15) << 2;
    *(v4bf*)(sQ + r * 80 + c) = *(const v4bf*)(qbase + (size_t)r * DHD + c);
  }

  int srow = tid >> 2, sq4 = tid & 3;
  int mrow = m0 + srow;
  float run_m = -3.0e38f, run_l = 0.f;
  bool rmask = im[b * TT + mrow] != 0;

  int i0 = wv * 2, i1 = i0 + 1;
  v8f o0 = {0.f,0.f,0.f,0.f,0.f,0.f,0.f,0.f};
  v8f o1 = o0;

  int nend = KVLEN;
  { int cc = m0 + 63 + TMEM + 1; if (cc < nend) nend = cc; }

  for (int n0 = 0; n0 < nend; n0 += 64) {
    __syncthreads();
    int jjbase = n0 + 960 - m0;   // jj = 1023 + n - m at (mrel=63, nrel=0)
    // ---- stage K, V^T, PE tiles ----
    for (int i = tid; i < 64 * 16; i += 256) {
      int r = i >> 4, c = (i & 15) << 2;
      *(v4bf*)(sK + r * 80 + c) = *(const v4bf*)(kbase + (size_t)(n0 + r) * DHD + c);
      v4bf vv = *(const v4bf*)(vbase + (size_t)(n0 + r) * DHD + c);
      sVt[(c + 0) * 80 + r] = vv[0];
      sVt[(c + 1) * 80 + r] = vv[1];
      sVt[(c + 2) * 80 + r] = vv[2];
      sVt[(c + 3) * 80 + r] = vv[3];
    }
    for (int i = tid; i < 128 * 16; i += 256) {
      int r = i >> 4, c = (i & 15) << 2;
      *(v4bf*)(sPE + r * 80 + c) = *(const v4bf*)(pebase + (size_t)(jjbase + r) * DHD + c);
    }
    if (n0 + 64 < nend)  // hint next K tile toward the caches
      __builtin_prefetch((const void*)(kbase + (size_t)(n0 + 64 + (tid & 63)) * DHD), 0, 1);
    __syncthreads();

    // ---- S (4x4 tiles) and R (4x8 tiles) via WMMA: 48 tiles over 8 waves ----
#pragma unroll
    for (int it = 0; it < 6; it++) {
      int idx = wv + (it << 3);
      const __bf16* Bbase;
      float* dst; int ld, tr, tc;
      if (idx < 16) { tr = idx >> 2; tc = idx & 3; Bbase = sK + tc * 16 * 80; dst = sS; ld = 64; }
      else { int jx = idx - 16; tr = jx >> 3; tc = jx & 7; Bbase = sPE + tc * 16 * 80; dst = sR; ld = 128; }
      const __bf16* Abase = sQ + tr * 16 * 80;
      v8f acc = {0.f,0.f,0.f,0.f,0.f,0.f,0.f,0.f};
      acc = wmma_bf16(frag_a(Abase, 80),      frag_b(Bbase, 80),      acc);
      acc = wmma_bf16(frag_a(Abase + 32, 80), frag_b(Bbase + 32, 80), acc);
      float* op = dst + (tr * 16) * ld + tc * 16 + lcol;
#pragma unroll
      for (int r = 0; r < 8; r++) op[(r + 8 * hf) * ld] = acc[r];
    }
    __syncthreads();

    // ---- online softmax: 4 threads per row, 16 cols each ----
    float l[16], p[16];
#pragma unroll
    for (int c = 0; c < 16; c++) {
      int nrel = (sq4 << 4) + c;
      int n = n0 + nrel;
      float val = (sS[srow * 64 + nrel] + sR[srow * 128 + nrel + 63 - srow]) * 0.125f;
      bool valid = (n < KVLEN) && ((n - mrow) <= TMEM);
      if (n >= TMEM) valid = valid && rmask && (im[b * TT + (n - TMEM)] != 0);
      l[c] = valid ? val : -1.0e30f;
    }
    float tmax = l[0];
#pragma unroll
    for (int c = 1; c < 16; c++) tmax = fmaxf(tmax, l[c]);
    tmax = fmaxf(tmax, __shfl_xor(tmax, 1, 32));
    tmax = fmaxf(tmax, __shfl_xor(tmax, 2, 32));
    float newm  = fmaxf(run_m, tmax);
    float alpha = __expf(run_m - newm);
    float sum = 0.f;
#pragma unroll
    for (int c = 0; c < 16; c++) { p[c] = __expf(l[c] - newm); sum += p[c]; }
    sum += __shfl_xor(sum, 1, 32);
    sum += __shfl_xor(sum, 2, 32);
    run_l = run_l * alpha + sum;
    run_m = newm;
#pragma unroll
    for (int c = 0; c < 16; c++) sP[srow * 80 + (sq4 << 4) + c] = f2bf(p[c]);
    if (sq4 == 0) { sAl[srow] = alpha; sL[srow] = run_l; }
    __syncthreads();

    // ---- O rescale + O += P @ V (2 tiles per wave) ----
    int tr0 = i0 >> 2, tc0 = i0 & 3, tr1 = i1 >> 2, tc1 = i1 & 3;
#pragma unroll
    for (int r = 0; r < 8; r++) {
      o0[r] *= sAl[tr0 * 16 + r + 8 * hf];
      o1[r] *= sAl[tr1 * 16 + r + 8 * hf];
    }
    o0 = wmma_bf16(frag_a(sP + tr0 * 16 * 80, 80),      frag_b(sVt + tc0 * 16 * 80, 80),      o0);
    o0 = wmma_bf16(frag_a(sP + tr0 * 16 * 80 + 32, 80), frag_b(sVt + tc0 * 16 * 80 + 32, 80), o0);
    o1 = wmma_bf16(frag_a(sP + tr1 * 16 * 80, 80),      frag_b(sVt + tc1 * 16 * 80, 80),      o1);
    o1 = wmma_bf16(frag_a(sP + tr1 * 16 * 80 + 32, 80), frag_b(sVt + tc1 * 16 * 80 + 32, 80), o1);
  }
  __syncthreads();

  // ---- finalize: divide by softmax denom, store bf16 head output [b, t, h*64+d] ----
  int tr0 = i0 >> 2, tc0 = i0 & 3, tr1 = i1 >> 2, tc1 = i1 & 3;
#pragma unroll
  for (int r = 0; r < 8; r++) {
    {
      int rl = tr0 * 16 + r + 8 * hf;
      int d  = tc0 * 16 + lcol;
      aout[((size_t)b * TT + m0 + rl) * EE + h * DHD + d] = f2bf(o0[r] / sL[rl]);
    }
    {
      int rl = tr1 * 16 + r + 8 * hf;
      int d  = tc1 * 16 + lcol;
      aout[((size_t)b * TT + m0 + rl) * EE + h * DHD + d] = f2bf(o1[r] / sL[rl]);
    }
  }
}

// ---------------------------------------------------------------------------
// Output projection: out[2048,1024] = Aout(bf16) @ Wo^T + bo, f32 result
// ---------------------------------------------------------------------------
__global__ void __launch_bounds__(256) proj_out_kernel(const __bf16* __restrict__ aout,
                                                       const float* __restrict__ Wo,
                                                       const float* __restrict__ bo,
                                                       float* __restrict__ out) {
  __shared__ __bf16 sA[64 * 40];
  __shared__ __bf16 sB[64 * 40];
  int tid = threadIdx.x;
  int m0 = blockIdx.x * 64, n0 = blockIdx.y * 64;
  int wv = tid >> 5, lane = tid & 31, hf = lane >> 4, col = lane & 15;
  int i0 = wv * 2, i1 = i0 + 1;
  v8f acc0 = {0.f,0.f,0.f,0.f,0.f,0.f,0.f,0.f};
  v8f acc1 = acc0;
  for (int k0 = 0; k0 < EE; k0 += 32) {
    for (int i = tid; i < 256; i += 256) {
      int r = i >> 2, c = (i & 3) << 3;
      *(v8bf*)(sA + r * 40 + c) = *(const v8bf*)(aout + (size_t)(m0 + r) * EE + k0 + c);
    }
    for (int i = tid; i < 512; i += 256) {
      int r = i >> 3, c = (i & 7) << 2;
      float4 fb = *(const float4*)(Wo + (size_t)(n0 + r) * EE + k0 + c);
      __bf16* db = sB + r * 40 + c;
      db[0] = f2bf(fb.x); db[1] = f2bf(fb.y); db[2] = f2bf(fb.z); db[3] = f2bf(fb.w);
    }
    __syncthreads();
    acc0 = wmma_bf16(frag_a(sA + (i0 >> 2) * 16 * 40, 40), frag_b(sB + (i0 & 3) * 16 * 40, 40), acc0);
    acc1 = wmma_bf16(frag_a(sA + (i1 >> 2) * 16 * 40, 40), frag_b(sB + (i1 & 3) * 16 * 40, 40), acc1);
    __syncthreads();
  }
#pragma unroll
  for (int r = 0; r < 8; r++) {
    {
      int mg = m0 + (i0 >> 2) * 16 + r + 8 * hf;
      int ng = n0 + (i0 & 3) * 16 + col;
      out[(size_t)mg * EE + ng] = acc0[r] + bo[ng];
    }
    {
      int mg = m0 + (i1 >> 2) * 16 + r + 8 * hf;
      int ng = n0 + (i1 & 3) * 16 + col;
      out[(size_t)mg * EE + ng] = acc1[r] + bo[ng];
    }
  }
}

// ---------------------------------------------------------------------------
extern "C" void kernel_launch(void* const* d_in, const int* in_sizes, int n_in,
                              void* d_out, int out_size, void* d_ws, size_t ws_size,
                              hipStream_t stream) {
  const float* x        = (const float*)d_in[0];
  const float* memp     = (const float*)d_in[1];
  const float* lmem     = (const float*)d_in[2];
  const float* pos_emb  = (const float*)d_in[3];
  const unsigned char* im = (const unsigned char*)d_in[4];
  const float* Wq       = (const float*)d_in[5];
  const float* Wkv      = (const float*)d_in[6];
  const float* Wo       = (const float*)d_in[7];
  const float* bo       = (const float*)d_in[8];
  const float* mem_kv   = (const float*)d_in[9];
  float* out = (float*)d_out;

  // workspace carve-up (~31.5 MB total)
  char* ws = (char*)d_ws;
  size_t off = 0;
  auto take = [&](size_t bytes) -> char* {
    char* p = ws + off;
    off += (bytes + 255) & ~(size_t)255;
    return p;
  };
  __bf16* qbf  = (__bf16*)take((size_t)BB * HH * TT * DHD * 2);
  __bf16* kbf  = (__bf16*)take((size_t)BB * HH * KVPAD * DHD * 2);
  __bf16* vbf  = (__bf16*)take((size_t)BB * HH * KVPAD * DHD * 2);
  __bf16* pebf = (__bf16*)take((size_t)HH * PEPAD * DHD * 2);
  __bf16* abf  = (__bf16*)take((size_t)BB * TT * EE * 2);
  (void)ws_size; (void)n_in; (void)in_sizes; (void)out_size;

  // 1. pad/convert pos_emb
  pe_pad_kernel<<<(HH * PEPAD * DHD + 255) / 256, 256, 0, stream>>>(pos_emb, pebf);
  // 2. zero K/V pad rows
  kv_pad_kernel<<<(BB * HH * (KVPAD - KVLEN) * DHD + 255) / 256, 256, 0, stream>>>(kbf, vbf);
  // 3. Q projection
  proj_q_kernel<<<dim3(32, 16), 256, 0, stream>>>(x, Wq, qbf);
  // 4. KV projection (gathered virtual concat, 4360 rows)
  proj_kv_kernel<<<dim3((BB * KVLEN + 63) / 64, 32), 256, 0, stream>>>(x, memp, lmem, mem_kv, Wkv, kbf, vbf);
  // 5. flash attention with relative-position shift
  size_t shmem = (size_t)(64*80 + 64*80 + 64*80 + 128*80 + 64*80) * 2
               + (size_t)(64*64 + 64*128 + 64 + 64) * 4;   // 111,104 B
  flash_attn_kernel<<<dim3(TT / 64, BB * HH), 256, shmem, stream>>>(qbf, kbf, vbf, pebf, im, abf);
  // 6. output projection + bias
  proj_out_kernel<<<dim3(32, 16), 256, 0, stream>>>(abf, Wo, bo, out);
}
